// CplxKernel_79267916415211
// MI455X (gfx1250) — compile-verified, matrix operand order checked by reading, not run on hardware
//
#include <hip/hip_runtime.h>
#include <hip/hip_bf16.h>

// ---------------------------------------------------------------------------
// Fused NNConv stack for gfx1250 (MI455X), round 2.
//   - Edge MLP GEMMs on v_wmma_f32_16x16x32_bf16 (f32 accumulate).
//   - 32-edge tile per block (2 WMMA M-tiles): halves L2 weight traffic per
//     edge vs 16-edge tiles; every B-fragment feeds two WMMAs.
//   - Stage-4 A-fragments hoisted out of the N-loop (8x fewer ds_loads).
//   - [32,1024] Wedge tile lives in LDS (66KB of the 320KB WGP budget);
//     Wedge never touches HBM (saves ~5GB of HBM traffic vs materializing).
//   - Weights converted once/layer to K-major bf16 -> per-lane fragments are
//     two contiguous 16B runs (global_load_b128), L2-resident across blocks.
// ---------------------------------------------------------------------------

typedef __attribute__((ext_vector_type(16))) __bf16 v16bf;
typedef __attribute__((ext_vector_type(8)))  float  v8f;

#define N_NODES 8000
#define N_EDGES 48000
#define TC      32
#define KW      128
#define EF      6
#define NLAYER  13
#define NHEAD   6
#define TILE_E  32              // edges per block = 2 WMMA M-tiles
#define LDH     (KW + 8)        // LDS row stride (halfs) for h buffers
#define LDWE    (TC * TC + 8)   // LDS row stride (halfs) for Wedge tile

// --- WMMA fragment loaders --------------------------------------------------
// 16-bit A layout (CDNA5 ISA): lane L -> M = m0 + (L&15); K base =
// (L&16)?8:0; half j<8 -> K=base+j ; half j>=8 -> K=base+16+(j-8).
// B assumed symmetric with N across lanes; B supplied K-major (transposed).

__device__ __forceinline__ v16bf loadA_lds(const __bf16 (*S)[LDH], int m0, int k0,
                                           int lane) {
  const int m  = m0 + (lane & 15);
  const int kb = k0 + ((lane & 16) ? 8 : 0);
  const __bf16* p = &S[m][kb];
  v16bf f;
#pragma unroll
  for (int j = 0; j < 8; ++j) f[j] = p[j];
#pragma unroll
  for (int j = 0; j < 8; ++j) f[8 + j] = p[16 + j];
  return f;
}

// Wt is [N][K] (K-major), ldk = K stride in elements.
__device__ __forceinline__ v16bf loadB_kmaj(const __bf16* __restrict__ Wt, int ldk,
                                            int k0, int n0, int lane) {
  const int n  = n0 + (lane & 15);
  const int kb = k0 + ((lane & 16) ? 8 : 0);
  const __bf16* p = Wt + (long)n * ldk + kb;
  v16bf f;
#pragma unroll
  for (int j = 0; j < 8; ++j) f[j] = p[j];
#pragma unroll
  for (int j = 0; j < 8; ++j) f[8 + j] = p[16 + j];
  return f;
}

// C epilogue store: VGPR r -> M = m0 + r + 8*(lane>=16), N = n0 + (lane&15).
__device__ __forceinline__ void store_c_prelu(__bf16 (*Sout)[LDH], const v8f& acc,
                                              const float* __restrict__ bias,
                                              float slope, int m0, int n0, int lane) {
  const int n  = n0 + (lane & 15);
  const int mb = m0 + ((lane & 16) ? 8 : 0);
  const float bv = bias[n];
#pragma unroll
  for (int r = 0; r < 8; ++r) {
    float v = acc[r] + bv;
    v = (v >= 0.f) ? v : slope * v;
    Sout[mb + r][n] = (__bf16)v;
  }
}

// Two-M-tile GEMM: C[0..31,n0:n0+16] = A(32x128) * B(128x16); B loaded once
// per k-chunk and shared by both M accumulators.
__device__ __forceinline__ void gemm2_tile_prelu(const __bf16 (*Sin)[LDH],
                                                 __bf16 (*Sout)[LDH],
                                                 const __bf16* __restrict__ Wt,
                                                 const float* __restrict__ bias,
                                                 float slope, int n0, int lane) {
  v8f acc0 = {}, acc1 = {};
#pragma unroll
  for (int k0 = 0; k0 < KW; k0 += 32) {
    v16bf b  = loadB_kmaj(Wt, KW, k0, n0, lane);
    v16bf a0 = loadA_lds(Sin, 0, k0, lane);
    v16bf a1 = loadA_lds(Sin, 16, k0, lane);
    acc0 = __builtin_amdgcn_wmma_f32_16x16x32_bf16(false, a0, false, b,
                                                   (short)0, acc0, false, false);
    acc1 = __builtin_amdgcn_wmma_f32_16x16x32_bf16(false, a1, false, b,
                                                   (short)0, acc1, false, false);
  }
  store_c_prelu(Sout, acc0, bias, slope, 0, n0, lane);
  store_c_prelu(Sout, acc1, bias, slope, 16, n0, lane);
}

// --- Fused edge kernel: edge MLP (WMMA) + per-edge matvec + scatter-add -----
__global__ __launch_bounds__(256) void edge_fused_kernel(
    const float* __restrict__ edgeattr, const int* __restrict__ edgeidx,
    const float* __restrict__ x,
    const float* __restrict__ eW1, const float* __restrict__ eb1, const float* __restrict__ ea1,
    const __bf16* __restrict__ w2t, const float* __restrict__ eb2, const float* __restrict__ ea2,
    const __bf16* __restrict__ w3t, const float* __restrict__ eb3, const float* __restrict__ ea3,
    const __bf16* __restrict__ w4t, const float* __restrict__ eb4,
    float* __restrict__ agg) {
  __shared__ __bf16 sA[TILE_E][LDH];
  __shared__ __bf16 sB[TILE_E][LDH];
  __shared__ __bf16 sW[TILE_E][LDWE];   // 32 x 1024 Wedge tile (bf16, ~66KB)
  __shared__ float  sx[TILE_E][TC];     // gathered x[src]
  __shared__ float  sattr[TILE_E][EF];
  __shared__ int    sidx[2][TILE_E];

  const int tid  = threadIdx.x;
  const int lane = tid & 31;
  const int wv   = tid >> 5;            // 8 waves (wave32)
  const int e0   = blockIdx.x * TILE_E;

  if (tid < TILE_E) {
    sidx[0][tid] = edgeidx[e0 + tid];            // src
    sidx[1][tid] = edgeidx[N_EDGES + e0 + tid];  // dst
  }
  if (tid < TILE_E * EF) {
    int e = tid / EF, k = tid % EF;
    sattr[e][k] = edgeattr[(e0 + e) * EF + k];
  }
  __syncthreads();

  const float a1 = ea1[0], a2 = ea2[0], a3 = ea3[0];

  // Stage 1: h1 = prelu(attr @ eW1 + eb1)  (K=6 -> plain VALU) -> sA
#pragma unroll
  for (int i = 0; i < 16; ++i) {
    int idx = tid + 256 * i;           // 4096 = 32 edges * 128 ch
    int e = idx >> 7, c = idx & 127;
    float acc = eb1[c];
#pragma unroll
    for (int k = 0; k < EF; ++k) acc += sattr[e][k] * eW1[k * KW + c];
    acc = (acc >= 0.f) ? acc : a1 * acc;
    sA[e][c] = (__bf16)acc;
  }
  // Gather x[src] for this tile (f32, 4KB)
#pragma unroll
  for (int i = 0; i < 4; ++i) {
    int idx = tid + 256 * i;           // 1024 = 32 * 32
    int e = idx >> 5, c = idx & 31;
    sx[e][c] = x[(long)sidx[0][e] * TC + c];
  }
  __syncthreads();

  // Stage 2: h2 = prelu(h1 @ W2 + b2)   -- one N-tile of 16 per wave, 2 M-tiles
  gemm2_tile_prelu(sA, sB, w2t, eb2, a2, wv * 16, lane);
  __syncthreads();

  // Stage 3: h3 = prelu(h2 @ W3 + b3)
  gemm2_tile_prelu(sB, sA, w3t, eb3, a3, wv * 16, lane);
  __syncthreads();

  // Stage 4: Wedge = h3 @ W4 + b4 -> sW.  N=1024: 8 N-tiles per wave.
  // A-fragments (2 M-tiles x 4 K-chunks) hoisted: loaded once, reused 8x.
  {
    v16bf a0k[4], a1k[4];
#pragma unroll
    for (int kc = 0; kc < 4; ++kc) {
      a0k[kc] = loadA_lds(sA, 0, kc * 32, lane);
      a1k[kc] = loadA_lds(sA, 16, kc * 32, lane);
    }
#pragma unroll
    for (int t = 0; t < 8; ++t) {
      const int n0 = wv * 128 + t * 16;
      v8f acc0 = {}, acc1 = {};
#pragma unroll
      for (int kc = 0; kc < 4; ++kc) {
        v16bf b = loadB_kmaj(w4t, KW, kc * 32, n0, lane);
        acc0 = __builtin_amdgcn_wmma_f32_16x16x32_bf16(false, a0k[kc], false, b,
                                                       (short)0, acc0, false, false);
        acc1 = __builtin_amdgcn_wmma_f32_16x16x32_bf16(false, a1k[kc], false, b,
                                                       (short)0, acc1, false, false);
      }
      const int n  = n0 + (lane & 15);
      const int mb = (lane & 16) ? 8 : 0;
      const float bv = eb4[n];
#pragma unroll
      for (int r = 0; r < 8; ++r) {
        sW[mb + r][n]      = (__bf16)(acc0[r] + bv);
        sW[16 + mb + r][n] = (__bf16)(acc1[r] + bv);
      }
    }
  }
  __syncthreads();

  // Stage 5: msg[e,o] = sum_i x_src[e,i] * Wedge[e, i*32+o]; scatter-add
#pragma unroll
  for (int i = 0; i < 4; ++i) {
    int idx = tid + 256 * i;           // 1024 = 32 * 32
    int e = idx >> 5, o = idx & 31;
    float m = 0.f;
#pragma unroll
    for (int c = 0; c < TC; ++c) m += sx[e][c] * (float)sW[e][c * TC + o];
    atomicAdd(&agg[(long)sidx[1][e] * TC + o], m);
  }
}

// --- Small helper kernels ---------------------------------------------------
__global__ void zero_kernel(float* p, int n) {
  int i = blockIdx.x * 256 + threadIdx.x;
  if (i < n) p[i] = 0.f;
}

__global__ void degree_kernel(const int* __restrict__ edgeidx, float* __restrict__ deg) {
  int e = blockIdx.x * 256 + threadIdx.x;
  if (e < N_EDGES) atomicAdd(&deg[edgeidx[N_EDGES + e]], 1.f);
}

// f32 [K,N] row-major -> bf16 [N,K] K-major (for WMMA B fragments)
__global__ void convert_transpose_kernel(const float* __restrict__ src,
                                         __bf16* __restrict__ dst, int K, int N) {
  int i = blockIdx.x * 256 + threadIdx.x;
  if (i < K * N) {
    int k = i / N, n = i % N;
    dst[(long)n * K + k] = (__bf16)src[i];
  }
}

__global__ __launch_bounds__(256) void encoder_kernel(
    const float* __restrict__ xxinc, const float* __restrict__ xxcord,
    const float* __restrict__ W1, const float* __restrict__ b1, const float* __restrict__ a1,
    const float* __restrict__ W2, const float* __restrict__ b2, const float* __restrict__ a2,
    const float* __restrict__ W3, const float* __restrict__ b3,
    float* __restrict__ xout) {
  __shared__ float sW1[EF * 8], sb1[8], sW2[8 * 16], sb2[16], sW3[16 * 32], sb3[32];
  int tid = threadIdx.x;
  for (int i = tid; i < EF * 8; i += 256) sW1[i] = W1[i];
  for (int i = tid; i < 8 * 16; i += 256) sW2[i] = W2[i];
  for (int i = tid; i < 16 * 32; i += 256) sW3[i] = W3[i];
  if (tid < 8)  sb1[tid] = b1[tid];
  if (tid < 16) sb2[tid] = b2[tid];
  if (tid < 32) sb3[tid] = b3[tid];
  __syncthreads();
  int n = blockIdx.x * 256 + tid;
  if (n >= N_NODES) return;
  const float s1 = a1[0], s2 = a2[0];
  float in[EF];
#pragma unroll
  for (int k = 0; k < 3; ++k) { in[k] = xxinc[n * 3 + k]; in[3 + k] = xxcord[n * 3 + k]; }
  float h1[8];
#pragma unroll
  for (int c = 0; c < 8; ++c) {
    float s = sb1[c];
#pragma unroll
    for (int k = 0; k < EF; ++k) s += in[k] * sW1[k * 8 + c];
    h1[c] = (s >= 0.f) ? s : s1 * s;
  }
  float h2[16];
#pragma unroll
  for (int c = 0; c < 16; ++c) {
    float s = sb2[c];
#pragma unroll
    for (int k = 0; k < 8; ++k) s += h1[k] * sW2[k * 16 + c];
    h2[c] = (s >= 0.f) ? s : s2 * s;
  }
#pragma unroll
  for (int o = 0; o < 32; ++o) {
    float s = sb3[o];
#pragma unroll
    for (int k = 0; k < 16; ++k) s += h2[k] * sW3[k * 32 + o];
    xout[(long)n * TC + o] = s;
  }
}

__global__ __launch_bounds__(256) void node_update_kernel(
    const float* __restrict__ xin, const float* __restrict__ agg,
    const float* __restrict__ deg, const float* __restrict__ root,
    const float* __restrict__ rbias, const float* __restrict__ oa,
    float* __restrict__ xout) {
  __shared__ float sroot[TC * TC], sb[TC];
  int tid = threadIdx.x;
  for (int i = tid; i < TC * TC; i += 256) sroot[i] = root[i];
  if (tid < TC) sb[tid] = rbias[tid];
  __syncthreads();
  int n = blockIdx.x * 256 + tid;
  if (n >= N_NODES) return;
  const float a = oa[0];
  const float d = fmaxf(deg[n], 1.f);
  float xv[TC];
#pragma unroll
  for (int i = 0; i < TC; ++i) xv[i] = xin[(long)n * TC + i];
#pragma unroll
  for (int o = 0; o < TC; ++o) {
    float s = sb[o] + agg[(long)n * TC + o] / d;
#pragma unroll
    for (int i = 0; i < TC; ++i) s += xv[i] * sroot[i * TC + o];
    xout[(long)n * TC + o] = (s >= 0.f) ? s : a * s;
  }
}

__global__ __launch_bounds__(256) void decoder_kernel(
    const float* __restrict__ x,
    const float* __restrict__ dW1, const float* __restrict__ db1, const float* __restrict__ da1,
    const float* __restrict__ dW2, const float* __restrict__ db2, const float* __restrict__ da2,
    const float* __restrict__ dW3, const float* __restrict__ db3,
    float* __restrict__ out) {
  __shared__ float sW1[NHEAD * TC * 16], sb1[NHEAD * 16], sW2[NHEAD * 16 * 8],
      sb2[NHEAD * 8], sW3[NHEAD * 8], sb3[NHEAD], sa1[NHEAD], sa2[NHEAD];
  int tid = threadIdx.x;
  for (int i = tid; i < NHEAD * TC * 16; i += 256) sW1[i] = dW1[i];
  for (int i = tid; i < NHEAD * 16; i += 256) sb1[i] = db1[i];
  for (int i = tid; i < NHEAD * 16 * 8; i += 256) sW2[i] = dW2[i];
  for (int i = tid; i < NHEAD * 8; i += 256) { sb2[i] = db2[i]; sW3[i] = dW3[i]; }
  if (tid < NHEAD) { sb3[tid] = db3[tid]; sa1[tid] = da1[tid]; sa2[tid] = da2[tid]; }
  __syncthreads();
  int n = blockIdx.x * 256 + tid;
  if (n >= N_NODES) return;
  float xv[TC];
#pragma unroll
  for (int i = 0; i < TC; ++i) xv[i] = x[(long)n * TC + i];
  for (int h = 0; h < NHEAD; ++h) {
    float h1[16];
#pragma unroll
    for (int c = 0; c < 16; ++c) {
      float s = sb1[h * 16 + c];
#pragma unroll
      for (int k = 0; k < TC; ++k) s += xv[k] * sW1[h * TC * 16 + k * 16 + c];
      h1[c] = (s >= 0.f) ? s : sa1[h] * s;
    }
    float h2[8];
#pragma unroll
    for (int c = 0; c < 8; ++c) {
      float s = sb2[h * 8 + c];
#pragma unroll
      for (int k = 0; k < 16; ++k) s += h1[k] * sW2[h * 16 * 8 + k * 8 + c];
      h2[c] = (s >= 0.f) ? s : sa2[h] * s;
    }
    float o = sb3[h];
#pragma unroll
    for (int k = 0; k < 8; ++k) o += h2[k] * sW3[h * 8 + k];
    out[(long)n * NHEAD + h] = o;
  }
}

// --- Host driver ------------------------------------------------------------
extern "C" void kernel_launch(void* const* d_in, const int* in_sizes, int n_in,
                              void* d_out, int out_size, void* d_ws, size_t ws_size,
                              hipStream_t stream) {
  // Inputs in pytree-leaf order of setup_inputs():
  const float* xxinc    = (const float*)d_in[0];
  const float* xxcord   = (const float*)d_in[1];
  const int*   edgeidx  = (const int*)d_in[2];
  const float* edgeattr = (const float*)d_in[3];
  const float* encW1 = (const float*)d_in[4];
  const float* encb1 = (const float*)d_in[5];
  const float* enca1 = (const float*)d_in[6];
  const float* encW2 = (const float*)d_in[7];
  const float* encb2 = (const float*)d_in[8];
  const float* enca2 = (const float*)d_in[9];
  const float* encW3 = (const float*)d_in[10];
  const float* encb3 = (const float*)d_in[11];
  const float* leW1  = (const float*)d_in[12];  // [L,6,128]
  const float* leb1  = (const float*)d_in[13];  // [L,128]
  const float* lea1  = (const float*)d_in[14];  // [L]
  const float* leW2  = (const float*)d_in[15];  // [L,128,128]
  const float* leb2  = (const float*)d_in[16];
  const float* lea2  = (const float*)d_in[17];
  const float* leW3  = (const float*)d_in[18];  // [L,128,128]
  const float* leb3  = (const float*)d_in[19];
  const float* lea3  = (const float*)d_in[20];
  const float* leW4  = (const float*)d_in[21];  // [L,128,1024]
  const float* leb4  = (const float*)d_in[22];  // [L,1024]
  const float* lroot = (const float*)d_in[23];  // [L,32,32]
  const float* lrb   = (const float*)d_in[24];  // [L,32]
  const float* loa   = (const float*)d_in[25];  // [L]
  const float* dW1 = (const float*)d_in[26];
  const float* db1 = (const float*)d_in[27];
  const float* da1 = (const float*)d_in[28];
  const float* dW2 = (const float*)d_in[29];
  const float* db2 = (const float*)d_in[30];
  const float* da2 = (const float*)d_in[31];
  const float* dW3 = (const float*)d_in[32];
  const float* db3 = (const float*)d_in[33];

  // Workspace carve-up (all 256B aligned)
  auto align256 = [](size_t v) { return (v + 255) & ~size_t(255); };
  char* ws = (char*)d_ws;
  float* deg = (float*)ws;  ws += align256(N_NODES * sizeof(float));
  float* x0  = (float*)ws;  ws += align256((size_t)N_NODES * TC * sizeof(float));
  float* x1  = (float*)ws;  ws += align256((size_t)N_NODES * TC * sizeof(float));
  float* agg = (float*)ws;  ws += align256((size_t)N_NODES * TC * sizeof(float));
  __bf16* w2t = (__bf16*)ws; ws += align256((size_t)KW * KW * sizeof(__bf16));
  __bf16* w3t = (__bf16*)ws; ws += align256((size_t)KW * KW * sizeof(__bf16));
  __bf16* w4t = (__bf16*)ws; ws += align256((size_t)KW * TC * TC * sizeof(__bf16));

  const int nb_nodes = (N_NODES + 255) / 256;
  const int nb_edges = (N_EDGES + 255) / 256;

  zero_kernel<<<nb_nodes, 256, 0, stream>>>(deg, N_NODES);
  degree_kernel<<<nb_edges, 256, 0, stream>>>(edgeidx, deg);
  encoder_kernel<<<nb_nodes, 256, 0, stream>>>(xxinc, xxcord, encW1, encb1, enca1,
                                               encW2, encb2, enca2, encW3, encb3, x0);

  float* xin = x0;
  float* xout = x1;
  for (int l = 0; l < NLAYER; ++l) {
    const float* eW1 = leW1 + (size_t)l * EF * KW;
    const float* eb1 = leb1 + (size_t)l * KW;
    const float* eW2 = leW2 + (size_t)l * KW * KW;
    const float* eb2 = leb2 + (size_t)l * KW;
    const float* eW3 = leW3 + (size_t)l * KW * KW;
    const float* eb3 = leb3 + (size_t)l * KW;
    const float* eW4 = leW4 + (size_t)l * KW * TC * TC;
    const float* eb4 = leb4 + (size_t)l * TC * TC;

    convert_transpose_kernel<<<(KW * KW + 255) / 256, 256, 0, stream>>>(eW2, w2t, KW, KW);
    convert_transpose_kernel<<<(KW * KW + 255) / 256, 256, 0, stream>>>(eW3, w3t, KW, KW);
    convert_transpose_kernel<<<(KW * TC * TC + 255) / 256, 256, 0, stream>>>(eW4, w4t, KW, TC * TC);
    zero_kernel<<<(N_NODES * TC + 255) / 256, 256, 0, stream>>>(agg, N_NODES * TC);

    edge_fused_kernel<<<N_EDGES / TILE_E, 256, 0, stream>>>(
        edgeattr, edgeidx, xin,
        eW1, eb1, lea1 + l,
        w2t, eb2, lea2 + l,
        w3t, eb3, lea3 + l,
        w4t, eb4, agg);

    node_update_kernel<<<nb_nodes, 256, 0, stream>>>(
        xin, agg, deg, lroot + (size_t)l * TC * TC, lrb + (size_t)l * TC, loa + l, xout);

    float* t = xin; xin = xout; xout = t;
  }

  decoder_kernel<<<nb_nodes, 256, 0, stream>>>(xin, dW1, db1, da1, dW2, db2, da2,
                                               dW3, db3, (float*)d_out);
}
